// MultiHeadLatentAttention_37632503447826
// MI455X (gfx1250) — compile-verified, matrix operand order checked by reading, not run on hardware
//
#include <hip/hip_runtime.h>
#include <hip/hip_bf16.h>

// ---------------------------------------------------------------------------
// MLA forward for MI455X (gfx1250): bf16 WMMA everywhere, f32 accumulate.
// Async global->LDS staging (ASYNCcnt) for GEMM B tiles and attention Q tiles.
// S=2048, D_MODEL=2048, N_HEAD=16, D_HEAD=128, D_ROPE=64, D_KV=512, D_Q=1024
// ---------------------------------------------------------------------------

#define S_LEN  2048
#define DMODEL 2048
#define NHEAD  16
#define DHEAD  128
#define DROPE  64
#define DKV    512
#define DQL    1024
#define DTOT   (DHEAD + DROPE)   // 192

typedef __bf16 bf16;
typedef __attribute__((ext_vector_type(16))) __bf16 v16bf;
typedef __attribute__((ext_vector_type(8)))  float  v8f;

// ---------------------------------------------------------------------------
// WMMA helper: D = A(16x32 bf16) * B(32x16 bf16) + C(16x16 f32)
// ---------------------------------------------------------------------------
__device__ __forceinline__ v8f wmma_bf16(v16bf a, v16bf b, v8f c) {
  return __builtin_amdgcn_wmma_f32_16x16x32_bf16(
      /*neg_a=*/false, a, /*neg_b=*/false, b,
      /*c_mod=*/(short)0, c, /*reuse_a=*/false, /*reuse_b=*/false);
}

// ---------------------------------------------------------------------------
// Fragment loader (ISA 7.12.2, 16-bit A-matrix 16x32):
//   lane L (0..15):  row M=L,    K = {0..7, 16..23}
//   lane L (16..31): row M=L-16, K = {8..15, 24..31}
// Two 16-byte loads per lane. Works for A from row-major [16 x K], and for B
// when B is available transposed (row-major [N x K]). Source may be LDS.
// ---------------------------------------------------------------------------
__device__ __forceinline__ v16bf load_frag(const bf16* base, int ld) {
  const int lane = threadIdx.x & 31;
  const bf16* p = base + (size_t)(lane & 15) * ld + ((lane >> 4) << 3);
  union { v16bf v; uint4 u[2]; } r;
  r.u[0] = *(const uint4*)(p);
  r.u[1] = *(const uint4*)(p + 16);
  return r.v;
}

// ---------------------------------------------------------------------------
// CDNA5 async global->LDS copy, 16 bytes per lane (VGLOBAL op 98, ASYNCcnt).
// ---------------------------------------------------------------------------
__device__ __forceinline__ void async_copy_b128(const void* gsrc, void* lds_dst) {
  unsigned ldsa = (unsigned)(unsigned long long)lds_dst;   // flat->LDS truncation
  asm volatile("global_load_async_to_lds_b128 %0, %1, off"
               :: "v"(ldsa), "v"(gsrc) : "memory");
}
__device__ __forceinline__ void async_wait0() {
  asm volatile("s_wait_asynccnt 0" ::: "memory");
}

// ---------------------------------------------------------------------------
// Elementwise f32 -> bf16 conversion
// ---------------------------------------------------------------------------
__global__ void convert_f32_to_bf16(const float* __restrict__ in,
                                    bf16* __restrict__ out, int n) {
  for (int i = blockIdx.x * blockDim.x + threadIdx.x; i < n;
       i += gridDim.x * blockDim.x)
    out[i] = (bf16)in[i];
}

// ---------------------------------------------------------------------------
// Transpose f32 [K x N] -> bf16 [N x K]  (LDS-tiled 32x32)
// ---------------------------------------------------------------------------
__global__ void transpose_f32_to_bf16(const float* __restrict__ in,
                                      bf16* __restrict__ out, int K, int N) {
  __shared__ float tile[32][33];
  const int k0 = blockIdx.y * 32, n0 = blockIdx.x * 32;
  const int tx = threadIdx.x, ty = threadIdx.y;   // block (32,8)
  #pragma unroll
  for (int i = ty; i < 32; i += 8)
    tile[i][tx] = in[(size_t)(k0 + i) * N + n0 + tx];
  __syncthreads();
  #pragma unroll
  for (int i = ty; i < 32; i += 8)
    out[(size_t)(n0 + i) * K + k0 + tx] = (bf16)tile[tx][i];
}

// ---------------------------------------------------------------------------
// Generic bf16 WMMA GEMM:  C[M x N] = A[M x K] * Bt[N x K]^T
// Block = 256 threads = 8 waves; wave w owns rows m0 = blkY*128 + w*16,
// cols n0 = blkX*64. The 64x32 B tile is staged in LDS per K step with
// async global->LDS loads (one b128 per thread), shared by all 8 waves.
// launch_bounds(...,1): spend VGPRs, not occupancy -> no scratch spills.
// ---------------------------------------------------------------------------
template <bool F32OUT>
__global__ __launch_bounds__(256, 1)
void gemm_bf16_wmma(const bf16* __restrict__ A,
                    const bf16* __restrict__ Bt,
                    void* __restrict__ C,
                    int M, int N, int K, int ldc) {
  __shared__ __align__(16) bf16 BS[64][32];       // 4 KB
  const int wave = threadIdx.x >> 5;
  const int m0 = blockIdx.y * 128 + wave * 16;
  const int n0 = blockIdx.x * 64;

  // this thread's slice of the cooperative B-tile copy: row, 8-elem chunk
  const int br = threadIdx.x >> 2;                // 0..63
  const int bc = (threadIdx.x & 3) * 8;           // 0,8,16,24

  v8f acc[4] = {};
  const bf16* pa = A + (size_t)m0 * K;
  const bf16* pb = Bt + (size_t)(n0 + br) * K + bc;

  for (int k0 = 0; k0 < K; k0 += 32) {
    // ---- stage B tile (64 rows x 32 k) into LDS, async -------------------
    async_copy_b128(pb + k0, &BS[br][bc]);
    async_wait0();
    __syncthreads();

    if (k0 + 64 <= K) {
      __builtin_prefetch(pa + k0 + 32, 0, 0);
      __builtin_prefetch(pb + k0 + 32, 0, 0);
    }
    v16bf a = load_frag(pa + k0, K);
    #pragma unroll
    for (int t = 0; t < 4; ++t) {
      v16bf b = load_frag(&BS[t * 16][0], 32);
      acc[t] = wmma_bf16(a, b, acc[t]);
    }
    __syncthreads();
  }

  const int lane = threadIdx.x & 31;
  const int hi = lane >> 4, nl = lane & 15;
  #pragma unroll
  for (int t = 0; t < 4; ++t) {
    #pragma unroll
    for (int v = 0; v < 8; ++v) {
      const int m = m0 + v + 8 * hi;
      const int n = n0 + t * 16 + nl;
      if (F32OUT)
        ((float*)C)[(size_t)m * ldc + n] = acc[t][v];
      else
        ((bf16*)C)[(size_t)m * ldc + n] = (bf16)acc[t][v];
    }
  }
}

// ---------------------------------------------------------------------------
// Assemble per-head q/k (content ++ RoPE(rope-part)) and transposed V.
//   qh,kh : [NHEAD][S][192]    vT : [NHEAD][128][S]
// grid (S, NHEAD), block 192.
// ---------------------------------------------------------------------------
__global__ void assemble_qkv(const bf16* __restrict__ qc, const bf16* __restrict__ qr,
                             const bf16* __restrict__ kc, const bf16* __restrict__ kr,
                             const bf16* __restrict__ vv,
                             bf16* __restrict__ qh, bf16* __restrict__ kh,
                             bf16* __restrict__ vT) {
  const int s = blockIdx.x, h = blockIdx.y, j = threadIdx.x;   // j < 192
  const size_t oidx = ((size_t)h * S_LEN + s) * DTOT + j;
  float qv, kv;
  if (j < DHEAD) {
    const size_t ii = (size_t)s * (NHEAD * DHEAD) + h * DHEAD + j;
    qv = (float)qc[ii];
    kv = (float)kc[ii];
    vT[((size_t)h * DHEAD + j) * S_LEN + s] = vv[ii];
  } else {
    const int jj = j - DHEAD;
    const int i = jj >> 1;
    const float inv = __powf(10000.f, -(float)(2 * i) / (float)DROPE);
    const float ang = (float)s * inv;
    const float c = __cosf(ang), sn = __sinf(ang);
    const size_t rb = (size_t)s * (NHEAD * DROPE) + h * DROPE + 2 * i;
    const float q1 = (float)qr[rb], q2 = (float)qr[rb + 1];
    const float k1 = (float)kr[rb], k2 = (float)kr[rb + 1];
    if ((jj & 1) == 0) { qv = q1 * c - q2 * sn; kv = k1 * c - k2 * sn; }
    else               { qv = q1 * sn + q2 * c; kv = k1 * sn + k2 * c; }
  }
  qh[oidx] = (bf16)qv;
  kh[oidx] = (bf16)kv;
}

// ---------------------------------------------------------------------------
// Flash-style causal attention.
// Block = 128 threads = 4 waves; wave w owns the 16-row q tile at
// q0 = blkY*64 + w*16 for head blkX. Q tile (16x192 bf16, 6 KB) is staged
// per-wave into LDS with async loads; Q fragments are re-read from LDS each
// K chunk. Waves are fully independent (private LDS slices, no barriers).
// launch_bounds(...,1): full VGPR budget -> no scratch spills.
// ---------------------------------------------------------------------------
__global__ __launch_bounds__(128, 1)
void mla_attention(const bf16* __restrict__ qh,
                   const bf16* __restrict__ kh,
                   const bf16* __restrict__ vT,
                   bf16* __restrict__ ao) {
  __shared__ __align__(16) bf16 QS[4][16][DTOT];  // 4 x 6 KB
  __shared__ __align__(16) bf16 P[4][16][32];     // 4 x 1 KB
  const int wave = threadIdx.x >> 5;
  const int lane = threadIdx.x & 31;
  const int h = blockIdx.x;
  const int q0 = blockIdx.y * 64 + wave * 16;
  const int hi = lane >> 4, nl = lane & 15;
  const float scale = 0.0721687836487f;           // 1/sqrt(192)

  // ---- async-stage this wave's q tile: 6144 B = 12 x (32 lanes x 16 B) ----
  {
    const char* src = (const char*)(qh + ((size_t)h * S_LEN + q0) * DTOT);
    char* dst = (char*)&QS[wave][0][0];
    #pragma unroll
    for (int c = 0; c < 12; ++c)
      async_copy_b128(src + c * 512 + lane * 16, dst + c * 512 + lane * 16);
    async_wait0();
  }

  v8f o[8] = {};
  float mrow[8], lrow[8];
  #pragma unroll
  for (int v = 0; v < 8; ++v) { mrow[v] = -3.0e38f; lrow[v] = 0.f; }

  const bf16* kbase = kh + (size_t)h * S_LEN * DTOT;
  const bf16* qs = &QS[wave][0][0];
  bf16* pw = &P[wave][0][0];

  for (int k0 = 0; k0 < q0 + 16; k0 += 32) {
    // ---- scores: two 16x16 tiles covering 32 keys (Q frags from LDS) ------
    v8f s0 = {}, s1 = {};
    #pragma unroll
    for (int d0 = 0; d0 < DTOT; d0 += 32) {
      v16bf a  = load_frag(qs + d0, DTOT);
      v16bf b0 = load_frag(kbase + (size_t)k0 * DTOT + d0, DTOT);
      v16bf b1 = load_frag(kbase + (size_t)(k0 + 16) * DTOT + d0, DTOT);
      s0 = wmma_bf16(a, b0, s0);
      s1 = wmma_bf16(a, b1, s1);
    }
    // ---- online softmax: row stats + P tile, then one accumulator pass ----
    float corr[8];
    #pragma unroll
    for (int v = 0; v < 8; ++v) {
      const int m = q0 + v + 8 * hi;
      float a0 = s0[v] * scale + ((k0 + nl)      <= m ? 0.f : -1.0e9f);
      float a1 = s1[v] * scale + ((k0 + 16 + nl) <= m ? 0.f : -1.0e9f);
      float mx = fmaxf(a0, a1);
      #pragma unroll
      for (int off = 1; off < 16; off <<= 1)
        mx = fmaxf(mx, __shfl_xor(mx, off, 32));
      const float mnew = fmaxf(mrow[v], mx);
      corr[v] = __expf(mrow[v] - mnew);
      const float p0 = __expf(a0 - mnew);
      const float p1 = __expf(a1 - mnew);
      float rs = p0 + p1;
      #pragma unroll
      for (int off = 1; off < 16; off <<= 1)
        rs += __shfl_xor(rs, off, 32);
      lrow[v] = lrow[v] * corr[v] + rs;
      mrow[v] = mnew;
      pw[(v + 8 * hi) * 32 + nl]      = (bf16)p0;
      pw[(v + 8 * hi) * 32 + nl + 16] = (bf16)p1;
    }
    #pragma unroll
    for (int t = 0; t < 8; ++t)
      #pragma unroll
      for (int v = 0; v < 8; ++v)
        o[t][v] *= corr[v];
    asm volatile("" ::: "memory");   // keep P stores before P fragment loads
    // ---- P(16x32) @ V(32x128) --------------------------------------------
    v16bf pa = load_frag(pw, 32);
    #pragma unroll
    for (int t = 0; t < 8; ++t) {
      v16bf vb = load_frag(vT + ((size_t)h * DHEAD + t * 16) * S_LEN + k0, S_LEN);
      o[t] = wmma_bf16(pa, vb, o[t]);
    }
    asm volatile("" ::: "memory");   // keep P loads before next-iter stores
  }

  // ---- finalize: divide by row-sum, write (s, h*128+d) bf16 ----------------
  #pragma unroll
  for (int t = 0; t < 8; ++t) {
    #pragma unroll
    for (int v = 0; v < 8; ++v) {
      const int m = q0 + v + 8 * hi;
      const int d = h * DHEAD + t * 16 + nl;
      ao[(size_t)m * (NHEAD * DHEAD) + d] = (bf16)(o[t][v] / lrow[v]);
    }
  }
}

// ---------------------------------------------------------------------------
// Host-side launch
// ---------------------------------------------------------------------------
extern "C" void kernel_launch(void* const* d_in, const int* in_sizes, int n_in,
                              void* d_out, int out_size, void* d_ws, size_t ws_size,
                              hipStream_t stream) {
  (void)in_sizes; (void)n_in; (void)out_size; (void)ws_size;

  const float* x        = (const float*)d_in[0];
  // d_in[1] = mask (causal; recomputed from indices in-kernel)
  const float* W_kvd    = (const float*)d_in[2];   // 2048 x 512
  const float* W_kup    = (const float*)d_in[3];   // 512  x 2048
  const float* W_vup    = (const float*)d_in[4];   // 512  x 2048
  const float* W_qd     = (const float*)d_in[5];   // 2048 x 1024
  const float* W_qup    = (const float*)d_in[6];   // 1024 x 2048
  const float* W_qr     = (const float*)d_in[7];   // 1024 x 1024
  const float* W_kr     = (const float*)d_in[8];   // 2048 x 1024
  const float* W_out    = (const float*)d_in[9];   // 2048 x 2048
  float* out            = (float*)d_out;

  // ---- workspace layout (bf16 elements) -----------------------------------
  bf16* ws = (bf16*)d_ws;
  size_t off = 0;
  auto alloc = [&](size_t n) { bf16* p = ws + off; off += n; return p; };
  bf16* xb     = alloc((size_t)S_LEN * DMODEL);
  bf16* wkvd_t = alloc((size_t)DKV * DMODEL);
  bf16* wkup_t = alloc((size_t)(NHEAD * DHEAD) * DKV);
  bf16* wvup_t = alloc((size_t)(NHEAD * DHEAD) * DKV);
  bf16* wqd_t  = alloc((size_t)DQL * DMODEL);
  bf16* wqup_t = alloc((size_t)(NHEAD * DHEAD) * DQL);
  bf16* wqr_t  = alloc((size_t)(NHEAD * DROPE) * DQL);
  bf16* wkr_t  = alloc((size_t)(NHEAD * DROPE) * DMODEL);
  bf16* wout_t = alloc((size_t)DMODEL * (NHEAD * DHEAD));
  bf16* dkv    = alloc((size_t)S_LEN * DKV);
  bf16* dq     = alloc((size_t)S_LEN * DQL);
  bf16* kc     = alloc((size_t)S_LEN * NHEAD * DHEAD);
  bf16* vv     = alloc((size_t)S_LEN * NHEAD * DHEAD);
  bf16* qc     = alloc((size_t)S_LEN * NHEAD * DHEAD);
  bf16* qr     = alloc((size_t)S_LEN * NHEAD * DROPE);
  bf16* kr     = alloc((size_t)S_LEN * NHEAD * DROPE);
  bf16* qhb    = alloc((size_t)NHEAD * S_LEN * DTOT);
  bf16* khb    = alloc((size_t)NHEAD * S_LEN * DTOT);
  bf16* vT     = alloc((size_t)NHEAD * DHEAD * S_LEN);
  bf16* ao     = alloc((size_t)S_LEN * NHEAD * DHEAD);

  // ---- stage 0: convert x, transpose+convert weights -----------------------
  convert_f32_to_bf16<<<4096, 256, 0, stream>>>(x, xb, S_LEN * DMODEL);
  auto T = [&](const float* in, bf16* o_, int K, int N) {
    transpose_f32_to_bf16<<<dim3(N / 32, K / 32), dim3(32, 8), 0, stream>>>(in, o_, K, N);
  };
  T(W_kvd, wkvd_t, DMODEL, DKV);
  T(W_kup, wkup_t, DKV, NHEAD * DHEAD);
  T(W_vup, wvup_t, DKV, NHEAD * DHEAD);
  T(W_qd,  wqd_t,  DMODEL, DQL);
  T(W_qup, wqup_t, DQL, NHEAD * DHEAD);
  T(W_qr,  wqr_t,  DQL, NHEAD * DROPE);
  T(W_kr,  wkr_t,  DMODEL, NHEAD * DROPE);
  T(W_out, wout_t, NHEAD * DHEAD, DMODEL);

  // ---- stage 1: projection GEMMs -------------------------------------------
  auto G = [&](const bf16* A, const bf16* Bt, bf16* C, int M, int N, int K) {
    gemm_bf16_wmma<false><<<dim3(N / 64, M / 128), 256, 0, stream>>>(A, Bt, (void*)C, M, N, K, N);
  };
  G(xb,  wkvd_t, dkv, S_LEN, DKV, DMODEL);            // x @ W_kv_down
  G(dkv, wkup_t, kc,  S_LEN, NHEAD * DHEAD, DKV);     // down_kv @ W_k_up
  G(dkv, wvup_t, vv,  S_LEN, NHEAD * DHEAD, DKV);     // down_kv @ W_v_up
  G(xb,  wqd_t,  dq,  S_LEN, DQL, DMODEL);            // x @ W_q_down
  G(dq,  wqup_t, qc,  S_LEN, NHEAD * DHEAD, DQL);     // down_q @ W_q_up
  G(dq,  wqr_t,  qr,  S_LEN, NHEAD * DROPE, DQL);     // down_q @ W_q_rope
  G(xb,  wkr_t,  kr,  S_LEN, NHEAD * DROPE, DMODEL);  // x @ W_k_rope

  // ---- stage 2: RoPE + head assembly ---------------------------------------
  assemble_qkv<<<dim3(S_LEN, NHEAD), DTOT, 0, stream>>>(qc, qr, kc, kr, vv,
                                                        qhb, khb, vT);

  // ---- stage 3: causal attention -------------------------------------------
  mla_attention<<<dim3(NHEAD, S_LEN / 64), 128, 0, stream>>>(qhb, khb, vT, ao);

  // ---- stage 4: output projection (f32 out) --------------------------------
  gemm_bf16_wmma<true><<<dim3(DMODEL / 64, S_LEN / 128), 256, 0, stream>>>(
      ao, wout_t, (void*)out, S_LEN, DMODEL, NHEAD * DHEAD, DMODEL);
}
